// GNNAgent_84834194031327
// MI455X (gfx1250) — compile-verified, matrix operand order checked by reading, not run on hardware
//
#include <hip/hip_runtime.h>
#include <hip/hip_bf16.h>

typedef __attribute__((ext_vector_type(16))) _Float16 v16h;
typedef __attribute__((ext_vector_type(8)))  _Float16 v8h;
typedef __attribute__((ext_vector_type(8)))  float    v8f;

#define HD   256   // hidden dim
#define DIN  128   // input dim
#define HC   128   // HEADS*C
#define CC   32    // per-head channels
#define NACT 20

// ---------------- utility kernels ----------------

__global__ void zero_u32(unsigned int* p, long long n) {
    long long i = (long long)blockIdx.x * blockDim.x + threadIdx.x;
    if (i < n) p[i] = 0u;
}

__global__ void f32_to_f16_k(const float* __restrict__ x, _Float16* __restrict__ y, long long n) {
    long long i = (long long)blockIdx.x * blockDim.x + threadIdx.x;
    if (i < n) y[i] = (_Float16)x[i];
}

// WT[n*K + k] = W[k*Nc + n]   (f32 [K,Nc] row-major -> f16 [Nc,K] row-major)
__global__ void transpose_f16_k(const float* __restrict__ W, _Float16* __restrict__ WT, int K, int Nc) {
    int i = blockIdx.x * blockDim.x + threadIdx.x;
    if (i >= K * Nc) return;
    int n = i / K, k = i % K;
    WT[i] = (_Float16)W[(size_t)k * Nc + n];
}

__device__ inline float wred_sum(float v) {
#pragma unroll
    for (int off = 16; off > 0; off >>= 1) v += __shfl_xor(v, off, 32);
    return v;
}

// ---------------- WMMA GEMM ----------------
// C[M,Nc] = A[M,K] * BT[Nc,K]^T + bias, f16 in / f32 out.
// One wave computes a 16 x (16*NT) strip: A fragment loaded once per K-step,
// NT independent accumulators -> NT back-to-back WMMAs per K-step.
// Fragment layout per CDNA5 ISA 7.12.2:
//   A 16x32 f16 : lane&15 = M row, lane>>4 picks K-halves {0..7,16..23} vs {8..15,24..31}
//   B 32x16 f16 : same pattern with lane&15 = N col (we hold B transposed)
//   C 16x16 f32 : VGPR r -> M = r + 8*(lane>>4), N = lane&15
template <int K, int NT>
__global__ void gemm_wmma(const _Float16* __restrict__ A, const _Float16* __restrict__ BT,
                          const float* __restrict__ bias, float* __restrict__ C,
                          int M, int Nc) {
    const int Ngroups = (Nc >> 4) / NT;
    int tile = blockIdx.x * (blockDim.x >> 5) + (threadIdx.x >> 5);
    if (tile >= (M >> 4) * Ngroups) return;       // wave-uniform: EXEC stays all-ones
    int mtile = tile / Ngroups;
    int nbase = (tile % Ngroups) * NT;            // first N-tile of this wave's strip
    int lane  = threadIdx.x & 31;
    int half  = lane >> 4;                        // 0 or 1
    int l15   = lane & 15;

    const _Float16* arow  = A  + (size_t)(mtile * 16 + l15) * K + half * 8;
    const _Float16* bbase = BT + (size_t)(nbase * 16 + l15) * K + half * 8;

    v8f acc[NT];
#pragma unroll
    for (int j = 0; j < NT; ++j)
#pragma unroll
        for (int i = 0; i < 8; ++i) acc[j][i] = 0.0f;

#pragma unroll
    for (int k0 = 0; k0 < K; k0 += 32) {
        v8h a0 = *(const v8h*)(arow + k0);
        v8h a1 = *(const v8h*)(arow + k0 + 16);
        v16h a;
#pragma unroll
        for (int i = 0; i < 8; ++i) { a[i] = a0[i]; a[8 + i] = a1[i]; }
#pragma unroll
        for (int j = 0; j < NT; ++j) {
            const _Float16* brow = bbase + (size_t)j * 16 * K + k0;
            v8h b0 = *(const v8h*)(brow);
            v8h b1 = *(const v8h*)(brow + 16);
            v16h b;
#pragma unroll
            for (int i = 0; i < 8; ++i) { b[i] = b0[i]; b[8 + i] = b1[i]; }
            acc[j] = __builtin_amdgcn_wmma_f32_16x16x32_f16(false, a, false, b,
                                                            (short)0, acc[j], false, false);
        }
    }

    int rbase = mtile * 16 + half * 8;
#pragma unroll
    for (int j = 0; j < NT; ++j) {
        int col  = (nbase + j) * 16 + l15;
        float bv = bias ? bias[col] : 0.0f;
#pragma unroll
        for (int r = 0; r < 8; ++r)
            C[(size_t)(rbase + r) * Nc + col] = acc[j][r] + bv;
    }
}

// ---------------- ReLU + LayerNorm (row of H, wave per row) -> f16 ----------------
__global__ void relu_ln_f16(const float* __restrict__ X, const float* __restrict__ g,
                            const float* __restrict__ be, _Float16* __restrict__ Yh,
                            int N, int H) {
    int wave = (blockIdx.x * blockDim.x + threadIdx.x) >> 5;
    int lane = threadIdx.x & 31;
    if (wave >= N) return;
    const float* row = X + (size_t)wave * H;
    const int per = H >> 5;                        // <= 8
    float vals[8];
    float s = 0.0f;
#pragma unroll
    for (int i = 0; i < 8; ++i) {
        if (i < per) {
            float v = row[lane + i * 32];
            v = v > 0.0f ? v : 0.0f;
            vals[i] = v;
            s += v;
        }
    }
    s = wred_sum(s);
    float mean = s / (float)H;
    float q = 0.0f;
#pragma unroll
    for (int i = 0; i < 8; ++i)
        if (i < per) { float d = vals[i] - mean; q += d * d; }
    q = wred_sum(q);
    float inv = rsqrtf(q / (float)H + 1e-5f);
    _Float16* yrow = Yh + (size_t)wave * H;
#pragma unroll
    for (int i = 0; i < 8; ++i) {
        if (i < per) {
            int cidx = lane + i * 32;
            yrow[cidx] = (_Float16)((vals[i] - mean) * inv * g[cidx] + be[cidx]);
        }
    }
}

// ---------------- edge phase ----------------
__device__ inline unsigned int ord_enc(float f) {
    unsigned int b = __float_as_uint(f);
    return (b & 0x80000000u) ? ~b : (b | 0x80000000u);
}
__device__ inline float ord_dec(unsigned int e) {
    return __uint_as_float((e & 0x80000000u) ? (e & 0x7FFFFFFFu) : ~e);
}

// logits[e,h] = sum_c lrelu(xl[src,h,c] + xr[dst,h,c]) * att[h,c]; atomic segment-max per dst
__global__ void edge_logits(const float* __restrict__ XL, const float* __restrict__ XR,
                            const int* __restrict__ src, const int* __restrict__ dst,
                            const float* __restrict__ att, float* __restrict__ LG,
                            unsigned int* __restrict__ mEnc, int E) {
    int wave = (blockIdx.x * blockDim.x + threadIdx.x) >> 5;
    int lane = threadIdx.x & 31;
    if (wave >= E) return;
    int s = src[wave], d = dst[wave];
    const float* xls = XL + (size_t)s * HC;
    const float* xrd = XR + (size_t)d * HC;
    __builtin_prefetch(xrd, 0, 0);                 // global_prefetch_b8
#pragma unroll
    for (int h = 0; h < 4; ++h) {
        float v = xls[h * CC + lane] + xrd[h * CC + lane];
        v = v > 0.0f ? v : 0.2f * v;               // leaky relu
        float p = wred_sum(v * att[h * CC + lane]);
        if (lane == 0) {
            LG[(size_t)wave * 4 + h] = p;
            atomicMax(&mEnc[(size_t)d * 4 + h], ord_enc(p));
        }
    }
}

// a = exp(logit - m[dst]); segment-sum into s[dst]
__global__ void edge_exp_sum(const int* __restrict__ dst, float* __restrict__ LG,
                             const unsigned int* __restrict__ mEnc, float* __restrict__ sSum,
                             int E) {
    long long idx = (long long)blockIdx.x * blockDim.x + threadIdx.x;
    if (idx >= (long long)E * 4) return;
    int e = (int)(idx >> 2), h = (int)(idx & 3);
    int d = dst[e];
    float m = ord_dec(mEnc[(size_t)d * 4 + h]);
    float a = __expf(LG[idx] - m);
    LG[idx] = a;
    atomicAdd(&sSum[(size_t)d * 4 + h], a);
}

// out[dst,h,c] += xl[src,h,c] * alpha[e,h]. One wave per edge: indices loaded once,
// alpha computed by lanes 0..3 and shfl-broadcast; 4 coalesced 32-lane atomic rows.
__global__ void edge_aggregate(const float* __restrict__ XL, const int* __restrict__ src,
                               const int* __restrict__ dst, const float* __restrict__ LG,
                               const float* __restrict__ sSum, float* __restrict__ OUT, int E) {
    int wave = (blockIdx.x * blockDim.x + threadIdx.x) >> 5;
    int lane = threadIdx.x & 31;
    if (wave >= E) return;
    int s = src[wave], d = dst[wave];
    float al = 0.0f;
    if (lane < 4)
        al = LG[(size_t)wave * 4 + lane] / (sSum[(size_t)d * 4 + lane] + 1e-16f);
    const float* xls = XL + (size_t)s * HC;
    float* od = OUT + (size_t)d * HC;
#pragma unroll
    for (int h = 0; h < 4; ++h) {
        float alpha = __shfl(al, h, 32);
        atomicAdd(&od[h * CC + lane], xls[h * CC + lane] * alpha);
    }
}

// h = mean over heads + residual
__global__ void node_combine(const float* __restrict__ OUT, const float* __restrict__ RES,
                             float* __restrict__ Hb, int N) {
    long long idx = (long long)blockIdx.x * blockDim.x + threadIdx.x;
    if (idx >= (long long)N * CC) return;
    int n = (int)(idx >> 5), c = (int)(idx & 31);
    const float* o = OUT + (size_t)n * HC;
    Hb[idx] = 0.25f * (o[c] + o[c + 32] + o[c + 64] + o[c + 96]) + RES[(size_t)n * CC + c];
}

// final head: ReLU -> LN(32) -> Linear(32->20). One wave per node, lane = channel.
__global__ void final_head(const float* __restrict__ Hb, const float* __restrict__ g3,
                           const float* __restrict__ be3, const float* __restrict__ Wq,
                           const float* __restrict__ bq, float* __restrict__ out, int N) {
    int wave = (blockIdx.x * blockDim.x + threadIdx.x) >> 5;
    int lane = threadIdx.x & 31;
    if (wave >= N) return;
    float v = Hb[(size_t)wave * CC + lane];
    v = v > 0.0f ? v : 0.0f;
    float mean = wred_sum(v) * (1.0f / 32.0f);
    float d = v - mean;
    float var = wred_sum(d * d) * (1.0f / 32.0f);
    float y = d * rsqrtf(var + 1e-5f) * g3[lane] + be3[lane];
    float acc = (lane < NACT) ? bq[lane] : 0.0f;
#pragma unroll
    for (int c = 0; c < 32; ++c) {
        float yc = __shfl(y, c, 32);
        if (lane < NACT) acc += yc * Wq[c * NACT + lane];
    }
    if (lane < NACT) out[(size_t)wave * NACT + lane] = acc;
}

// ---------------- host ----------------

static inline int cdiv(long long a, long long b) { return (int)((a + b - 1) / b); }

extern "C" void kernel_launch(void* const* d_in, const int* in_sizes, int n_in,
                              void* d_out, int out_size, void* d_ws, size_t ws_size,
                              hipStream_t stream) {
    const float* inputs = (const float*)d_in[0];
    const int*   eidx   = (const int*)d_in[1];
    const float* W1 = (const float*)d_in[2];   const float* b1 = (const float*)d_in[3];
    const float* g1 = (const float*)d_in[4];   const float* be1= (const float*)d_in[5];
    const float* W2 = (const float*)d_in[6];   const float* b2 = (const float*)d_in[7];
    const float* g2 = (const float*)d_in[8];   const float* be2= (const float*)d_in[9];
    const float* Wl = (const float*)d_in[10];  const float* bl = (const float*)d_in[11];
    const float* Wr = (const float*)d_in[12];  const float* br = (const float*)d_in[13];
    const float* att= (const float*)d_in[14];
    const float* Wres=(const float*)d_in[15];  const float* bg = (const float*)d_in[16];
    const float* g3 = (const float*)d_in[17];  const float* be3= (const float*)d_in[18];
    const float* Wq = (const float*)d_in[19];  const float* bq = (const float*)d_in[20];
    float* out = (float*)d_out;

    const int N = in_sizes[0] / DIN;     // 50000 (multiple of 16)
    const int E = in_sizes[1] / 2;       // 800000
    const int* src = eidx;
    const int* dst = eidx + E;

    // workspace carve-out
    size_t off = 0;
    auto carve = [&](size_t bytes) -> char* {
        char* p = (char*)d_ws + off;
        off += (bytes + 255) & ~(size_t)255;
        return p;
    };
    _Float16* XinH  = (_Float16*)carve((size_t)N * DIN * 2);
    _Float16* W1T   = (_Float16*)carve((size_t)DIN * HD * 2);
    _Float16* W2T   = (_Float16*)carve((size_t)HD * HD * 2);
    _Float16* WlT   = (_Float16*)carve((size_t)HD * HC * 2);
    _Float16* WrT   = (_Float16*)carve((size_t)HD * HC * 2);
    _Float16* WresT = (_Float16*)carve((size_t)HD * CC * 2);
    float*    PRE   = (float*)   carve((size_t)N * HD * 4);   // reused for both encoder GEMMs
    _Float16* X1H   = (_Float16*)carve((size_t)N * HD * 2);
    _Float16* X2H   = (_Float16*)carve((size_t)N * HD * 2);
    float*    XL    = (float*)   carve((size_t)N * HC * 4);
    float*    XR    = (float*)   carve((size_t)N * HC * 4);
    float*    RES   = (float*)   carve((size_t)N * CC * 4);
    float*    LG    = (float*)   carve((size_t)E * 4 * 4);
    unsigned* MENC  = (unsigned*)carve((size_t)N * 4 * 4);
    float*    SSUM  = (float*)   carve((size_t)N * 4 * 4);
    float*    OUTA  = (float*)   carve((size_t)N * HC * 4);
    float*    HB    = (float*)   carve((size_t)N * CC * 4);

    const int T = 256;
    const int WPB = 4;            // waves per block in GEMM (128 threads)

    // zero accumulators (harness does not re-poison between replays)
    zero_u32<<<cdiv((long long)N * HC, T), T, 0, stream>>>((unsigned*)OUTA, (long long)N * HC);
    zero_u32<<<cdiv((long long)N * 4,  T), T, 0, stream>>>(MENC,            (long long)N * 4);
    zero_u32<<<cdiv((long long)N * 4,  T), T, 0, stream>>>((unsigned*)SSUM, (long long)N * 4);

    // precision prep: inputs + weights to f16, weights transposed to [Nc,K]
    f32_to_f16_k<<<cdiv((long long)N * DIN, T), T, 0, stream>>>(inputs, XinH, (long long)N * DIN);
    transpose_f16_k<<<cdiv(DIN * HD, T), T, 0, stream>>>(W1,   W1T,   DIN, HD);
    transpose_f16_k<<<cdiv(HD * HD,  T), T, 0, stream>>>(W2,   W2T,   HD,  HD);
    transpose_f16_k<<<cdiv(HD * HC,  T), T, 0, stream>>>(Wl,   WlT,   HD,  HC);
    transpose_f16_k<<<cdiv(HD * HC,  T), T, 0, stream>>>(Wr,   WrT,   HD,  HC);
    transpose_f16_k<<<cdiv(HD * CC,  T), T, 0, stream>>>(Wres, WresT, HD,  CC);

    // encoder layer 1: [N,128]x[128,256], 16x64 strip per wave
    {
        int tiles = (N >> 4) * ((HD >> 4) / 4);
        gemm_wmma<DIN, 4><<<cdiv(tiles, WPB), WPB * 32, 0, stream>>>(XinH, W1T, b1, PRE, N, HD);
        relu_ln_f16<<<cdiv(N, 8), 256, 0, stream>>>(PRE, g1, be1, X1H, N, HD);
    }
    // encoder layer 2: [N,256]x[256,256]
    {
        int tiles = (N >> 4) * ((HD >> 4) / 4);
        gemm_wmma<HD, 4><<<cdiv(tiles, WPB), WPB * 32, 0, stream>>>(X1H, W2T, b2, PRE, N, HD);
        relu_ln_f16<<<cdiv(N, 8), 256, 0, stream>>>(PRE, g2, be2, X2H, N, HD);
    }
    // GAT transforms + residual: xl, xr [N,128]; res [N,32]
    {
        int tiles_l = (N >> 4) * ((HC >> 4) / 4);
        gemm_wmma<HD, 4><<<cdiv(tiles_l, WPB), WPB * 32, 0, stream>>>(X2H, WlT, bl, XL, N, HC);
        gemm_wmma<HD, 4><<<cdiv(tiles_l, WPB), WPB * 32, 0, stream>>>(X2H, WrT, br, XR, N, HC);
        int tiles_r = (N >> 4) * ((CC >> 4) / 2);
        gemm_wmma<HD, 2><<<cdiv(tiles_r, WPB), WPB * 32, 0, stream>>>(X2H, WresT, bg, RES, N, CC);
    }
    // edge attention (memory/atomic-bound phase)
    edge_logits<<<cdiv(E, 8), 256, 0, stream>>>(XL, XR, src, dst, att, LG, MENC, E);
    edge_exp_sum<<<cdiv((long long)E * 4, T), T, 0, stream>>>(dst, LG, MENC, SSUM, E);
    edge_aggregate<<<cdiv(E, 8), 256, 0, stream>>>(XL, src, dst, LG, SSUM, OUTA, E);
    // head
    node_combine<<<cdiv((long long)N * CC, T), T, 0, stream>>>(OUTA, RES, HB, N);
    final_head<<<cdiv(N, 8), 256, 0, stream>>>(HB, g3, be3, Wq, bq, out, N);

    (void)n_in; (void)out_size; (void)ws_size;
}
